// GCN_LSTM_84542136255050
// MI455X (gfx1250) — matrix operand order repaired and checked by reading, not compiled, not run on hardware
//
#include <hip/hip_runtime.h>
#include <hip/hip_bf16.h>

// ---------------------------------------------------------------------------
// GCN(128->128) + scatter-norm aggregate + LSTM(H=128) + FC(128->32)
// MI455X / gfx1250: wave32, WMMA f32_16x16x32_f16 (fp16 in, fp32 accumulate)
// ---------------------------------------------------------------------------

typedef __attribute__((ext_vector_type(16))) _Float16 v16h;
typedef __attribute__((ext_vector_type(8)))  _Float16 v8h;
typedef __attribute__((ext_vector_type(8)))  float    v8f;

#define NNODES 50000
#define FDIM   128
#define HDIM   128
#define TSTEPS 9
#define ODIM   32
#define APAD   136  // 16x128 f16 A-tile row stride in LDS (bank-conflict pad)

// -------------------- WMMA helpers (CDNA5 fragment layouts) ----------------

__device__ __forceinline__ v8f wmma_f16(v16h a, v16h b, v8f c) {
  // (neg_a, A, neg_b, B, c_mod, C, reuse_a, reuse_b)
  return __builtin_amdgcn_wmma_f32_16x16x32_f16(false, a, false, b, (short)0, c,
                                                false, false);
}

// A fragment 16x32 (f16) built from f32 row-major source.
// lane holds row M = m0 + (lane&15); half i -> K = k0 + i + 8*hi (+8 for i>=8)
__device__ __forceinline__ v16h load_a_f32(const float* __restrict__ base,
                                           int ld, int m0, int k0, int lane) {
  const int hi = lane >> 4;
  const float* p = base + (size_t)(m0 + (lane & 15)) * ld + k0 + 8 * hi;
  float4 x0 = *(const float4*)(p + 0);
  float4 x1 = *(const float4*)(p + 4);
  float4 x2 = *(const float4*)(p + 16);
  float4 x3 = *(const float4*)(p + 20);
  v16h a;
  a[0]  = (_Float16)x0.x; a[1]  = (_Float16)x0.y;
  a[2]  = (_Float16)x0.z; a[3]  = (_Float16)x0.w;
  a[4]  = (_Float16)x1.x; a[5]  = (_Float16)x1.y;
  a[6]  = (_Float16)x1.z; a[7]  = (_Float16)x1.w;
  a[8]  = (_Float16)x2.x; a[9]  = (_Float16)x2.y;
  a[10] = (_Float16)x2.z; a[11] = (_Float16)x2.w;
  a[12] = (_Float16)x3.x; a[13] = (_Float16)x3.y;
  a[14] = (_Float16)x3.z; a[15] = (_Float16)x3.w;
  return a;
}

// A fragment from an f16 LDS tile staged with row stride APAD halves.
// Same K mapping as above; two aligned 16B LDS reads per lane.
__device__ __forceinline__ v16h load_a_lds(const _Float16* abuf, int k0,
                                           int lane) {
  const int hi = lane >> 4;
  const _Float16* p = abuf + (lane & 15) * APAD + k0 + 8 * hi;
  v8h lo = *(const v8h*)(p);       // K = k0+8*hi .. +7
  v8h hh = *(const v8h*)(p + 16);  // K = k0+16+8*hi .. +7
  return __builtin_shufflevector(lo, hh, 0, 1, 2, 3, 4, 5, 6, 7,
                                 8, 9, 10, 11, 12, 13, 14, 15);
}

// B fragment 32x16 (f16) from column-major f16 weights (bcol[n][k], k contig).
// lane holds column N = n0 + (lane&15); half i -> K = k0 + i + 8*hi (+8 for i>=8)
__device__ __forceinline__ v16h load_b_f16(const _Float16* __restrict__ bcol,
                                           int ldk, int n0, int k0, int lane) {
  const int hi = lane >> 4;
  const _Float16* p = bcol + (size_t)(n0 + (lane & 15)) * ldk + k0 + 8 * hi;
  v8h lo = *(const v8h*)(p);
  v8h hh = *(const v8h*)(p + 16);
  return __builtin_shufflevector(lo, hh, 0, 1, 2, 3, 4, 5, 6, 7,
                                 8, 9, 10, 11, 12, 13, 14, 15);
}

__device__ __forceinline__ float sigf(float x) {
  return 1.0f / (1.0f + __expf(-x));
}

// -------------------- preprocessing kernels --------------------------------

// Convert all weights to f16 (B column-major layouts) + combine LSTM biases.
__global__ __launch_bounds__(256) void prep_kernel(
    const float* __restrict__ Wg, const float* __restrict__ Wih,
    const float* __restrict__ Whh, const float* __restrict__ Wfc,
    const float* __restrict__ bih, const float* __restrict__ bhh,
    _Float16* __restrict__ Wg_t, _Float16* __restrict__ Wih_h,
    _Float16* __restrict__ Whh_h, _Float16* __restrict__ Wfc_h,
    float* __restrict__ biasc) {
  int idx = blockIdx.x * 256 + threadIdx.x;
  if (idx < FDIM * HDIM) {  // transpose: Wg is [F_IN][H], need [n=H][k=F_IN]
    int f = idx >> 7, h = idx & 127;
    Wg_t[h * FDIM + f] = (_Float16)Wg[f * HDIM + h];
  }
  if (idx < 4 * HDIM * HDIM) {  // W_ih/W_hh are [512][128] = B col-major already
    Wih_h[idx] = (_Float16)Wih[idx];
    Whh_h[idx] = (_Float16)Whh[idx];
  }
  if (idx < ODIM * HDIM) {  // W_fc is [32][128] = B col-major already
    Wfc_h[idx] = (_Float16)Wfc[idx];
  }
  if (idx < 4 * HDIM) biasc[idx] = bih[idx] + bhh[idx];
}

__global__ __launch_bounds__(256) void degree_kernel(
    const int* __restrict__ col, float* __restrict__ deg, int E) {
  int e = blockIdx.x * 256 + threadIdx.x;
  if (e < E) atomicAdd(&deg[col[e]], 1.0f);
}

__global__ __launch_bounds__(256) void dinv_kernel(
    const float* __restrict__ deg, float* __restrict__ dinv, int n) {
  int i = blockIdx.x * 256 + threadIdx.x;
  if (i < n) dinv[i] = rsqrtf(deg[i] + 1.0f);  // +1 = self loop
}

// -------------------- GCN dense part: XW = x_t @ W_gcn ---------------------
// 1 wave per block; each wave computes a 16x128 output strip with 8 WMMA
// accumulators, K swept in 4 steps of 32.
__global__ __launch_bounds__(32) void gcn_xw_kernel(
    const float* __restrict__ x, const _Float16* __restrict__ Wt,
    float* __restrict__ xw) {
  const int lane = threadIdx.x;
  const int m0 = blockIdx.x * 16;
  v8f acc[8] = {};
#pragma unroll
  for (int kk = 0; kk < 4; ++kk) {
    const int k0 = kk * 32;
    v16h a = load_a_f32(x, FDIM, m0, k0, lane);
#pragma unroll
    for (int nt = 0; nt < 8; ++nt) {
      v16h b = load_b_f16(Wt, FDIM, nt * 16, k0, lane);
      acc[nt] = wmma_f16(a, b, acc[nt]);
    }
  }
  const int hi = lane >> 4, n = lane & 15;
#pragma unroll
  for (int nt = 0; nt < 8; ++nt)
#pragma unroll
    for (int r = 0; r < 8; ++r)
      xw[(size_t)(m0 + r + 8 * hi) * HDIM + nt * 16 + n] = acc[nt][r];
}

// -------------------- edge scatter (L2-resident atomics) -------------------
// 1 wave per edge; each lane gathers float4 of XW[row], scales by norm,
// atomically adds into AGG[col].
__global__ __launch_bounds__(256) void scatter_kernel(
    const float* __restrict__ xw, const int* __restrict__ row,
    const int* __restrict__ col, const float* __restrict__ dinv,
    float* __restrict__ agg, int E) {
  long long tid = (long long)blockIdx.x * 256 + threadIdx.x;
  int e = (int)(tid >> 5);
  if (e >= E) return;
  int lane = (int)(tid & 31);
  int r = row[e], c = col[e];
  float norm = dinv[r] * dinv[c];
  float4 v = *((const float4*)(xw + (size_t)r * HDIM) + lane);
  float* dst = agg + (size_t)c * HDIM + lane * 4;
  atomicAdd(dst + 0, v.x * norm);
  atomicAdd(dst + 1, v.y * norm);
  atomicAdd(dst + 2, v.z * norm);
  atomicAdd(dst + 3, v.w * norm);
}

// Fold self-loop term + bias + relu, in place on agg (becomes emb).
__global__ __launch_bounds__(256) void relu_bias_kernel(
    float* __restrict__ agg, const float* __restrict__ xw,
    const float* __restrict__ dinv, const float* __restrict__ bg, int total) {
  int idx = blockIdx.x * 256 + threadIdx.x;
  if (idx >= total) return;
  int i = idx >> 7, j = idx & 127;
  float d = dinv[i];
  float v = agg[idx] + xw[idx] * d * d + bg[j];
  agg[idx] = v > 0.0f ? v : 0.0f;
}

// -------------------- fused LSTM step --------------------------------------
// Block = 256 threads (8 waves), 16 rows. Both A tiles (emb, h) are staged
// ONCE into LDS as f16 (converted during staging), so the 8 waves build A
// fragments with two ds_load_b128 each instead of 8x redundant global loads
// + converts. gates[16][512] accumulates via WMMA into LDS, then the
// recurrence update runs in the same block — gates never hit HBM.
__global__ __launch_bounds__(256) void lstm_step_kernel(
    const float* __restrict__ emb, const _Float16* __restrict__ Wih,
    const _Float16* __restrict__ Whh, const float* __restrict__ biasc,
    float* __restrict__ hbuf, float* __restrict__ cbuf) {
  __shared__ __align__(16) _Float16 a_lds[2][16 * APAD];  // [0]=emb, [1]=h
  __shared__ float gates[16 * 512];
  const int lane = threadIdx.x & 31;
  const int wid = threadIdx.x >> 5;  // 8 waves; wave handles 64 gate cols
  const int m0 = blockIdx.x * 16;
  const int nbase = wid * 64;

  // ---- stage A tiles (f32 -> f16) once per block ----
#pragma unroll
  for (int s = threadIdx.x; s < 16 * HDIM; s += 256) {
    int r = s >> 7, k = s & 127;
    size_t off = (size_t)(m0 + r) * HDIM + k;
    a_lds[0][r * APAD + k] = (_Float16)emb[off];
    a_lds[1][r * APAD + k] = (_Float16)hbuf[off];
  }
  __syncthreads();

  // ---- gates = emb @ Wih^T + h @ Whh^T ----
  v8f acc[4] = {};
#pragma unroll
  for (int kk = 0; kk < 4; ++kk) {
    const int k0 = kk * 32;
    v16h a = load_a_lds(a_lds[0], k0, lane);
#pragma unroll
    for (int q = 0; q < 4; ++q) {
      v16h b = load_b_f16(Wih, HDIM, nbase + q * 16, k0, lane);
      acc[q] = wmma_f16(a, b, acc[q]);
    }
  }
#pragma unroll
  for (int kk = 0; kk < 4; ++kk) {
    const int k0 = kk * 32;
    v16h a = load_a_lds(a_lds[1], k0, lane);
#pragma unroll
    for (int q = 0; q < 4; ++q) {
      v16h b = load_b_f16(Whh, HDIM, nbase + q * 16, k0, lane);
      acc[q] = wmma_f16(a, b, acc[q]);
    }
  }
  const int hi = lane >> 4, n = lane & 15;
#pragma unroll
  for (int q = 0; q < 4; ++q)
#pragma unroll
    for (int r = 0; r < 8; ++r)
      gates[(r + 8 * hi) * 512 + nbase + q * 16 + n] = acc[q][r];
  __syncthreads();

  // ---- pointwise recurrence: 16 rows x 128 cols = 2048 elems / 256 thr ----
#pragma unroll
  for (int s = threadIdx.x; s < 16 * HDIM; s += 256) {
    int r = s >> 7, j = s & 127;
    size_t off = (size_t)(m0 + r) * HDIM + j;
    float ig = gates[r * 512 + j] + biasc[j];
    float fg = gates[r * 512 + 128 + j] + biasc[128 + j];
    float gg = gates[r * 512 + 256 + j] + biasc[256 + j];
    float og = gates[r * 512 + 384 + j] + biasc[384 + j];
    float c = sigf(fg) * cbuf[off] + sigf(ig) * tanhf(gg);
    float h = sigf(og) * tanhf(c);
    cbuf[off] = c;
    hbuf[off] = h;
  }
}

// -------------------- FC head: out[:, t, :] = h @ W_fc^T + b ---------------
__global__ __launch_bounds__(32) void fc_kernel(
    const float* __restrict__ h, const _Float16* __restrict__ Wfc,
    const float* __restrict__ bfc, float* __restrict__ out, int t) {
  const int lane = threadIdx.x;
  const int m0 = blockIdx.x * 16;
  v8f acc0 = {}, acc1 = {};
#pragma unroll
  for (int kk = 0; kk < 4; ++kk) {
    const int k0 = kk * 32;
    v16h a = load_a_f32(h, HDIM, m0, k0, lane);
    acc0 = wmma_f16(a, load_b_f16(Wfc, HDIM, 0, k0, lane), acc0);
    acc1 = wmma_f16(a, load_b_f16(Wfc, HDIM, 16, k0, lane), acc1);
  }
  const int hi = lane >> 4, n = lane & 15;
  const float b0 = bfc[n], b1 = bfc[16 + n];
#pragma unroll
  for (int r = 0; r < 8; ++r) {
    size_t rowoff = (size_t)(m0 + r + 8 * hi) * (TSTEPS * ODIM) + (size_t)t * ODIM;
    out[rowoff + n] = acc0[r] + b0;
    out[rowoff + 16 + n] = acc1[r] + b1;
  }
}

// -------------------- launch -----------------------------------------------

extern "C" void kernel_launch(void* const* d_in, const int* in_sizes, int n_in,
                              void* d_out, int out_size, void* d_ws,
                              size_t ws_size, hipStream_t stream) {
  (void)n_in; (void)out_size; (void)ws_size;
  const float* x_seq = (const float*)d_in[0];
  const int*   edge  = (const int*)d_in[1];
  const float* W_gcn = (const float*)d_in[2];
  const float* b_gcn = (const float*)d_in[3];
  const float* W_ih  = (const float*)d_in[4];
  const float* W_hh  = (const float*)d_in[5];
  const float* b_ih  = (const float*)d_in[6];
  const float* b_hh  = (const float*)d_in[7];
  const float* W_fc  = (const float*)d_in[8];
  const float* b_fc  = (const float*)d_in[9];
  float* out = (float*)d_out;

  const int N = NNODES;
  const int E = in_sizes[1] / 2;
  const int* row = edge;
  const int* col = edge + E;

  // workspace carve-out (256B aligned)
  char* wp = (char*)d_ws;
  auto carve = [&](size_t bytes) {
    char* p = wp;
    wp += (bytes + 255) & ~(size_t)255;
    return p;
  };
  float*     deg   = (float*)carve((size_t)N * 4);
  float*     dinv  = (float*)carve((size_t)N * 4);
  _Float16*  Wg_t  = (_Float16*)carve((size_t)FDIM * HDIM * 2);
  _Float16*  Wih_h = (_Float16*)carve((size_t)4 * HDIM * HDIM * 2);
  _Float16*  Whh_h = (_Float16*)carve((size_t)4 * HDIM * HDIM * 2);
  _Float16*  Wfc_h = (_Float16*)carve((size_t)ODIM * HDIM * 2);
  float*     biasc = (float*)carve((size_t)4 * HDIM * 4);
  float*     xw    = (float*)carve((size_t)N * HDIM * 4);
  float*     agg   = (float*)carve((size_t)N * HDIM * 4);  // becomes emb
  float*     hbuf  = (float*)carve((size_t)N * HDIM * 4);
  float*     cbuf  = (float*)carve((size_t)N * HDIM * 4);

  hipMemsetAsync(deg, 0, (size_t)N * 4, stream);
  hipMemsetAsync(hbuf, 0, (size_t)N * HDIM * 4, stream);
  hipMemsetAsync(cbuf, 0, (size_t)N * HDIM * 4, stream);

  prep_kernel<<<(4 * HDIM * HDIM + 255) / 256, 256, 0, stream>>>(
      W_gcn, W_ih, W_hh, W_fc, b_ih, b_hh, Wg_t, Wih_h, Whh_h, Wfc_h, biasc);
  degree_kernel<<<(E + 255) / 256, 256, 0, stream>>>(col, deg, E);
  dinv_kernel<<<(N + 255) / 256, 256, 0, stream>>>(deg, dinv, N);

  const int mtiles = N / 16;  // 3125 (exact): full waves only -> EXEC all-1s
  const long long sthreads = (long long)E * 32;
  const unsigned sblocks = (unsigned)((sthreads + 255) / 256);

  for (int t = 0; t < TSTEPS; ++t) {
    hipMemsetAsync(agg, 0, (size_t)N * HDIM * 4, stream);
    gcn_xw_kernel<<<mtiles, 32, 0, stream>>>(
        x_seq + (size_t)t * N * FDIM, Wg_t, xw);
    scatter_kernel<<<sblocks, 256, 0, stream>>>(xw, row, col, dinv, agg, E);
    relu_bias_kernel<<<(N * HDIM + 255) / 256, 256, 0, stream>>>(
        agg, xw, dinv, b_gcn, N * HDIM);
    lstm_step_kernel<<<mtiles, 256, 0, stream>>>(
        agg, Wih_h, Whh_h, biasc, hbuf, cbuf);
    fc_kernel<<<mtiles, 32, 0, stream>>>(hbuf, Wfc_h, b_fc, out, t);
  }
}
